// DynamicPrototypeInteraction_19842748907621
// MI455X (gfx1250) — matrix operand order repaired and checked by reading, not compiled, not run on hardware
//
#include <hip/hip_runtime.h>
#include <hip/hip_bf16.h>
#include <math.h>

typedef __attribute__((ext_vector_type(8)))  float  v8f;
typedef __attribute__((ext_vector_type(8)))  __bf16 v8bf;
typedef __attribute__((ext_vector_type(16))) __bf16 v16bf;

#define C_DIM 256
#define KD    128
#define P_DIM 256
#define HW    4096
#define H_DIM 64
#define W_DIM 64
#define B_DIM 16
#define TILE  64          // pixel rows per block (4 waves x 16 rows)
#define LDX   264         // stride for klen=256 tiles (256 + 8 pad -> conflict-free banks)
#define LQX   136         // stride for klen=128 tiles (128 + 8 pad)
#define SW_ELEMS 34816    // shared weight buffer: max(128*264, 256*136) bf16

// ---------------- WMMA helpers (CDNA5 16x16x32 bf16, f32 accumulate) ----------------

// A-fragment (16x32 bf16) from row-major LDS tile.
// ISA 7.12.2: lanes 0-15 hold M=lane, K = {kb..kb+7, kb+16..kb+23};
//             lanes 16-31 hold M=lane-16, K = {kb+8..kb+15, kb+24..kb+31}.
__device__ __forceinline__ v16bf make_a(const __bf16* base, int ld, int m, int half, int kb) {
  const __bf16* p = base + m * ld + kb + half * 8;
  union { v16bf v; v8bf h[2]; } u;
  u.h[0] = *(const v8bf*)(p);
  u.h[1] = *(const v8bf*)(p + 16);
  return u.v;
}

// B-fragment (32x16 bf16) from an N-major, column-padded LDS matrix (elem (k,n) at n*ldp+k).
// lanes 0-15: col N=lane, K = {kb..kb+7, kb+16..kb+23}; lanes 16-31: col N=lane-16, K = {kb+8..kb+15, kb+24..kb+31}.
__device__ __forceinline__ v16bf make_b(const __bf16* B, int ldp, int n, int half, int kb) {
  const __bf16* p = B + (size_t)n * ldp + kb + half * 8;
  union { v16bf v; v8bf h[2]; } u;
  u.h[0] = *(const v8bf*)(p);
  u.h[1] = *(const v8bf*)(p + 16);
  return u.v;
}

__device__ __forceinline__ v8f wmma_bf16(v16bf a, v16bf b, v8f c) {
  return __builtin_amdgcn_wmma_f32_16x16x32_bf16(false, a, false, b, (short)0, c, false, false);
}

// One fused GEMM phase: acc[NB] (16x16 blocks) += A[16 x KLEN] * B[KLEN x NB*16].
// Software-pipelined: the next group's 8 B fragments (and next A) are prefetched into a
// second register set BEFORE the current group's 8 WMMAs issue, so DS latency overlaps
// the XDL work instead of a per-WMMA s_wait_dscnt 0 drain. Fully unrolled (KLEN is a
// template arg) so the double-buffer copies disappear via register renaming.
template <int NB, int KLEN>
__device__ __forceinline__ void gemm_phase(const __bf16* As, int lda,
                                           const __bf16* Bs, int ldb,
                                           int lm, int half, v8f* acc) {
  v16bf a_cur = make_a(As, lda, lm, half, 0);
  v16bf bcur[8], bnext[8];
  #pragma unroll
  for (int i = 0; i < 8; ++i) bcur[i] = make_b(Bs, ldb, i * 16 + lm, half, 0);

  #pragma unroll
  for (int k = 0; k < KLEN; k += 32) {
    if (NB == 8) {
      v16bf a_nxt = a_cur;
      if (k + 32 < KLEN) {
        a_nxt = make_a(As, lda, lm, half, k + 32);
        #pragma unroll
        for (int i = 0; i < 8; ++i) bnext[i] = make_b(Bs, ldb, i * 16 + lm, half, k + 32);
      }
      #pragma unroll
      for (int i = 0; i < 8; ++i) acc[i] = wmma_bf16(a_cur, bcur[i], acc[i]);
      a_cur = a_nxt;
      #pragma unroll
      for (int i = 0; i < 8; ++i) bcur[i] = bnext[i];
    } else {
      // group 1 prefetch overlaps group 0 WMMAs
      #pragma unroll
      for (int i = 0; i < 8; ++i) bnext[i] = make_b(Bs, ldb, (8 + i) * 16 + lm, half, k);
      #pragma unroll
      for (int i = 0; i < 8; ++i) acc[i] = wmma_bf16(a_cur, bcur[i], acc[i]);
      // next k-step group 0 prefetch overlaps group 1 WMMAs
      v16bf a_nxt = a_cur;
      if (k + 32 < KLEN) {
        a_nxt = make_a(As, lda, lm, half, k + 32);
        #pragma unroll
        for (int i = 0; i < 8; ++i) bcur[i] = make_b(Bs, ldb, i * 16 + lm, half, k + 32);
      }
      #pragma unroll
      for (int i = 0; i < 8; ++i) acc[8 + i] = wmma_bf16(a_cur, bnext[i], acc[8 + i]);
      a_cur = a_nxt;
    }
  }
}

// ---------------- CDNA5 async global->LDS copy (ASYNCcnt-tracked DMA path) ----------------
// Per-lane: LDS[vdst_lane] <- MEM[vaddr_lane], 16B. One instruction moves 512B per wave.
__device__ __forceinline__ void async_copy_16B(unsigned lds_off, const void* gsrc) {
  asm volatile("global_load_async_to_lds_b128 %0, %1, off"
               :: "v"(lds_off), "v"(gsrc) : "memory");
}
__device__ __forceinline__ void wait_async_all() {
  asm volatile("s_wait_asynccnt 0x0" ::: "memory");
}

// Stage an N-major weight matrix [ncols x klen] (dense in global) into LDS with
// column padding (ldp > klen) for bank-conflict-free B-fragment reads.
// Low 32 bits of a generic LDS pointer are the LDS byte offset (ISA 10.2 aperture rule).
__device__ __forceinline__ void stage_w(__bf16* dst, const __bf16* __restrict__ src,
                                        int ncols, int klen, int ldp) {
  const int chunks = klen >> 3;   // 16B chunks per column
  for (int col = threadIdx.x; col < ncols; col += 128) {
    unsigned lbase = (unsigned)(size_t)(dst + (size_t)col * ldp);
    const __bf16* g = src + (size_t)col * klen;
    #pragma unroll 4
    for (int ch = 0; ch < chunks; ++ch)
      async_copy_16B(lbase + ch * 16, g + ch * 8);
  }
}

// ---------------- prep: bf16 weight copies ----------------
__global__ void prep_weights(const float* __restrict__ qw, const float* __restrict__ ow,
                             __bf16* __restrict__ qwb, __bf16* __restrict__ owb) {
  int i = blockIdx.x * blockDim.x + threadIdx.x;
  if (i < KD * C_DIM) qwb[i] = (__bf16)qw[i];   // q_w [KD,C] row-major == B-ready (n=kd, k=c)
  if (i < C_DIM * KD) owb[i] = (__bf16)ow[i];   // o_w [C,KD] row-major == B-ready (n=c, k=d)
}

// ---------------- prep: k = l2norm(m @ k_w^T + k_b), vT = (m @ v_w^T + v_b)^T ----------------
__global__ __launch_bounds__(128) void prep_kv(
    const float* __restrict__ m, const float* __restrict__ kw, const float* __restrict__ kb,
    const float* __restrict__ vw, const float* __restrict__ vb,
    __bf16* __restrict__ kbf, __bf16* __restrict__ vtbf) {
  __shared__ float mrow[KD];
  __shared__ float red[KD];
  int p = blockIdx.x, d = threadIdx.x;
  mrow[d] = m[(size_t)p * KD + d];
  __syncthreads();
  float kv = 0.f, vv = 0.f;
  #pragma unroll 4
  for (int j = 0; j < KD; ++j) {
    float mj = mrow[j];
    kv = fmaf(mj, kw[(size_t)d * KD + j], kv);
    vv = fmaf(mj, vw[(size_t)d * KD + j], vv);
  }
  kv += kb[d]; vv += vb[d];
  red[d] = kv * kv;
  __syncthreads();
  for (int s = 64; s > 0; s >>= 1) { if (d < s) red[d] += red[d + s]; __syncthreads(); }
  float inv = 1.f / fmaxf(sqrtf(red[0]), 1e-12f);
  kbf[(size_t)p * KD + d]     = (__bf16)(kv * inv);  // [P,KD] == B-ready (n=p, k=d)
  vtbf[(size_t)d * P_DIM + p] = (__bf16)vv;          // v^T [KD,P] == B-ready (n=d, k=p)
}

// ---------------- fused: q-proj -> l2norm -> logits -> softmax -> attn*V -> o-proj ----------------
__global__ __launch_bounds__(128) void attn_fused(
    const float* __restrict__ x, const __bf16* __restrict__ qw, const __bf16* __restrict__ kbf,
    const __bf16* __restrict__ vtbf, const __bf16* __restrict__ ow,
    const float* __restrict__ ob, const float* __restrict__ qb,
    float* __restrict__ mid) {
  __shared__ __bf16 s_w[SW_ELEMS];      // block-shared weight buffer, reused each GEMM phase
  __shared__ __bf16 s_x[4][16 * LDX];   // per-wave x tile, reused for attn weights
  __shared__ __bf16 s_q[4][16 * LQX];   // per-wave q stage, reused for hidden stage
  const int wave = threadIdx.x >> 5;
  const int lane = threadIdx.x & 31;
  const int half = lane >> 4;
  const int lm   = lane & 15;
  const int tile = blockIdx.x;
  const int b    = tile / (HW / TILE);
  const int hw0  = (tile % (HW / TILE)) * TILE + wave * 16;   // this wave's 16 pixels
  __bf16* xs = s_x[wave];
  __bf16* qs = s_q[wave];
  const v8f vzero = {0.f,0.f,0.f,0.f,0.f,0.f,0.f,0.f};

  // ---- prologue: async-stage q_w (n=128 cols, k=256) while staging the x strip ----
  stage_w(s_w, qw, KD, C_DIM, LDX);

  // x strip [16 rows x 256 ch], NCHW -> bf16 LDS row-major; float4 along hw
  const float* xb = x + (size_t)b * C_DIM * HW + hw0;
  for (int idx = lane; idx < 4 * C_DIM; idx += 32) {
    int r4 = (idx & 3) * 4;
    int c  = idx >> 2;
    float4 v = *(const float4*)(xb + (size_t)c * HW + r4);
    xs[(r4 + 0) * LDX + c] = (__bf16)v.x;
    xs[(r4 + 1) * LDX + c] = (__bf16)v.y;
    xs[(r4 + 2) * LDX + c] = (__bf16)v.z;
    xs[(r4 + 3) * LDX + c] = (__bf16)v.w;
  }
  wait_async_all();
  __syncthreads();

  // ---- GEMM 1: q[16x128] = x[16x256] * q_w^T (B from LDS) ----
  v8f accq[8];
  #pragma unroll
  for (int nb = 0; nb < 8; ++nb) accq[nb] = vzero;
  gemm_phase<8, C_DIM>(xs, LDX, s_w, LDX, lm, half, accq);
  __syncthreads();                       // all waves done reading s_w(q_w)
  stage_w(s_w, kbf, P_DIM, KD, LQX);     // prefetch k (n=256, k=128) under epilogue work

  float qbv[8];
  #pragma unroll
  for (int nb = 0; nb < 8; ++nb) qbv[nb] = qb[nb * 16 + lm];

  // +bias, row l2norm (rows 0-7 in lanes 0-15, rows 8-15 in lanes 16-31)
  #pragma unroll
  for (int j = 0; j < 8; ++j) {
    float vq[8]; float ss = 0.f;
    #pragma unroll
    for (int nb = 0; nb < 8; ++nb) { float t = accq[nb][j] + qbv[nb]; vq[nb] = t; ss += t * t; }
    #pragma unroll
    for (int msk = 1; msk < 16; msk <<= 1) ss += __shfl_xor(ss, msk, 16);
    float inv = 1.f / fmaxf(sqrtf(ss), 1e-12f);
    int row = j + half * 8;
    #pragma unroll
    for (int nb = 0; nb < 8; ++nb) qs[row * LQX + nb * 16 + lm] = (__bf16)(vq[nb] * inv);
  }
  wait_async_all();
  __syncthreads();

  // ---- GEMM 2: logits[16x256] = q[16x128] * k^T ----
  v8f accl[16];
  #pragma unroll
  for (int nb = 0; nb < 16; ++nb) accl[nb] = vzero;
  gemm_phase<16, KD>(qs, LQX, s_w, LQX, lm, half, accl);
  __syncthreads();                       // all waves done reading s_w(k)
  stage_w(s_w, vtbf, KD, P_DIM, LDX);    // prefetch v^T (n=128, k=256) under softmax

  // softmax over P=256 of clamp(logit/T, +-30); attn -> bf16 into reused x region
  const float invT = 1.f / 0.07f;
  #pragma unroll
  for (int j = 0; j < 8; ++j) {
    float vals[16]; float mx = -1e30f;
    #pragma unroll
    for (int nb = 0; nb < 16; ++nb) {
      float v = fminf(fmaxf(accl[nb][j] * invT, -30.f), 30.f);
      vals[nb] = v; mx = fmaxf(mx, v);
    }
    #pragma unroll
    for (int msk = 1; msk < 16; msk <<= 1) mx = fmaxf(mx, __shfl_xor(mx, msk, 16));
    float sum = 0.f;
    #pragma unroll
    for (int nb = 0; nb < 16; ++nb) { float e = __expf(vals[nb] - mx); vals[nb] = e; sum += e; }
    #pragma unroll
    for (int msk = 1; msk < 16; msk <<= 1) sum += __shfl_xor(sum, msk, 16);
    float is = 1.f / sum;
    int row = j + half * 8;
    #pragma unroll
    for (int nb = 0; nb < 16; ++nb) xs[row * LDX + nb * 16 + lm] = (__bf16)(vals[nb] * is);
  }
  wait_async_all();
  __syncthreads();

  // ---- GEMM 3: hid[16x128] = attn[16x256] * v ----
  v8f acco[8];
  #pragma unroll
  for (int nb = 0; nb < 8; ++nb) acco[nb] = vzero;
  gemm_phase<8, P_DIM>(xs, LDX, s_w, LDX, lm, half, acco);
  __syncthreads();                       // all waves done reading s_w(v^T)
  stage_w(s_w, ow, C_DIM, KD, LQX);      // prefetch o_w (n=256, k=128) under hid staging

  #pragma unroll
  for (int j = 0; j < 8; ++j) {
    int row = j + half * 8;
    #pragma unroll
    for (int nb = 0; nb < 8; ++nb) qs[row * LQX + nb * 16 + lm] = (__bf16)acco[nb][j];
  }
  wait_async_all();
  __syncthreads();

  // ---- GEMM 4: out[16x256] = hid[16x128] * o_w^T ; store fp32 NHWC intermediate ----
  v8f accp[16];
  #pragma unroll
  for (int nb = 0; nb < 16; ++nb) accp[nb] = vzero;
  gemm_phase<16, KD>(qs, LQX, s_w, LQX, lm, half, accp);

  float* midb = mid + ((size_t)b * HW + hw0) * C_DIM;
  #pragma unroll
  for (int j = 0; j < 8; ++j) {
    int row = j + half * 8;
    #pragma unroll
    for (int nb = 0; nb < 16; ++nb) {
      int c = nb * 16 + lm;
      midb[(size_t)row * C_DIM + c] = accp[nb][j] + ob[c];  // 16 lanes -> 64B coalesced
    }
  }
}

// ---------------- epilogue: depthwise 3x3 + BN1 + GELU(exact) + add + residual + BN2 ----------------
__global__ __launch_bounds__(256) void conv_epilogue(
    const float* __restrict__ x, const float* __restrict__ mid, const float* __restrict__ dw,
    const float* __restrict__ bn1w, const float* __restrict__ bn1b,
    const float* __restrict__ bn2w, const float* __restrict__ bn2b,
    const float* __restrict__ gam, float* __restrict__ out) {
  int blk = blockIdx.x;
  int b = blk >> 12;            // / 4096
  int hw = blk & 4095;
  int h = hw >> 6, w = hw & 63;
  int c = threadIdx.x;
  const float* midb = mid + (size_t)b * HW * C_DIM;
  float acc = 0.f;
  #pragma unroll
  for (int dy = -1; dy <= 1; ++dy) {
    int h2 = h + dy; if (h2 < 0 || h2 >= H_DIM) continue;
    #pragma unroll
    for (int dx = -1; dx <= 1; ++dx) {
      int w2 = w + dx; if (w2 < 0 || w2 >= W_DIM) continue;
      acc = fmaf(midb[(size_t)(h2 * W_DIM + w2) * C_DIM + c],
                 dw[c * 9 + (dy + 1) * 3 + (dx + 1)], acc);   // NHWC: coalesced along c
    }
  }
  const float bnScale = 0.9999950000374997f;  // 1/sqrt(1 + 1e-5)
  float z = acc * (bn1w[c] * bnScale) + bn1b[c];
  float g = 0.5f * z * (1.f + erff(z * 0.7071067811865475f)); // exact GELU
  float o = midb[(size_t)hw * C_DIM + c] + g;
  float res = x[((size_t)b * C_DIM + c) * HW + hw];
  out[((size_t)b * C_DIM + c) * HW + hw] =
      (res + o * gam[c]) * (bn2w[c] * bnScale) + bn2b[c];
}

// ---------------- host ----------------
extern "C" void kernel_launch(void* const* d_in, const int* in_sizes, int n_in,
                              void* d_out, int out_size, void* d_ws, size_t ws_size,
                              hipStream_t stream) {
  const float* x    = (const float*)d_in[0];
  const float* mfe  = (const float*)d_in[1];
  const float* q_w  = (const float*)d_in[2];
  const float* q_b  = (const float*)d_in[3];
  const float* k_w  = (const float*)d_in[4];
  const float* k_b  = (const float*)d_in[5];
  const float* v_w  = (const float*)d_in[6];
  const float* v_b  = (const float*)d_in[7];
  const float* o_w  = (const float*)d_in[8];
  const float* o_b  = (const float*)d_in[9];
  const float* dw   = (const float*)d_in[10];
  const float* bn1w = (const float*)d_in[11];
  const float* bn1b = (const float*)d_in[12];
  const float* bn2w = (const float*)d_in[13];
  const float* bn2b = (const float*)d_in[14];
  const float* gam  = (const float*)d_in[15];

  char* ws = (char*)d_ws;
  __bf16* qwb = (__bf16*)(ws);                 // 64 KiB  q_w bf16 (B-ready)
  __bf16* owb = (__bf16*)(ws + (64 << 10));    // 64 KiB  o_w bf16 (B-ready)
  __bf16* kbf = (__bf16*)(ws + (128 << 10));   // 64 KiB  k    bf16 (B-ready)
  __bf16* vtb = (__bf16*)(ws + (192 << 10));   // 64 KiB  v^T  bf16 (B-ready)
  float*  mid = (float*)(ws + (256 << 10));    // 64 MiB  attention output, NHWC fp32
  float*  out = (float*)d_out;

  prep_weights<<<(KD * C_DIM + 255) / 256, 256, 0, stream>>>(q_w, o_w, qwb, owb);
  prep_kv<<<P_DIM, KD, 0, stream>>>(mfe, k_w, k_b, v_w, v_b, kbf, vtb);
  attn_fused<<<(B_DIM * HW) / TILE, 128, 0, stream>>>(x, qwb, kbf, vtb, owb, o_b, q_b, mid);
  conv_epilogue<<<B_DIM * HW, 256, 0, stream>>>(x, mid, dw, bn1w, bn1b, bn2w, bn2b, gam, out);
}
